// CausalSelfAttention_45122926411822
// MI455X (gfx1250) — matrix operand order repaired and checked by reading, not compiled, not run on hardware
//
#include <hip/hip_runtime.h>

#define DEVINL __device__ __forceinline__

typedef __attribute__((ext_vector_type(16))) __bf16 v16bf;
typedef __attribute__((ext_vector_type(8)))  float   v8f;
typedef __attribute__((ext_vector_type(4)))  unsigned int u32x4;

union Frag16 { v16bf bf; u32x4 u[2]; };

constexpr int kB = 4, kT = 2048, kC = 1024, kH = 16, kD = 64;
constexpr int kM = kB * kT;        // 8192 tokens
constexpr int kNqkv = 3 * kC;      // 3072

DEVINL v8f wmma_bf16(v16bf a, v16bf b, v8f c) {
  return __builtin_amdgcn_wmma_f32_16x16x32_bf16(false, a, false, b, (short)0, c,
                                                 false, false);
}
DEVINL u32x4 ld16(const __bf16* p) { return *(const u32x4*)p; }

// ---------------------------------------------------------------- converters
__global__ void cvt_f32_bf16(const float* __restrict__ s, __bf16* __restrict__ d,
                             long long n) {
  long long i = (long long)blockIdx.x * blockDim.x + threadIdx.x;
  long long st = (long long)gridDim.x * blockDim.x;
  for (; i < n; i += st) d[i] = (__bf16)s[i];
}

// src [K][N] fp32 row-major  ->  dst [N][K] bf16 row-major
__global__ void transpose_f32_bf16(const float* __restrict__ s, __bf16* __restrict__ d,
                                   int K, int N) {
  long long n = (long long)K * N;
  long long i = (long long)blockIdx.x * blockDim.x + threadIdx.x;
  long long st = (long long)gridDim.x * blockDim.x;
  for (; i < n; i += st) {
    int k = (int)(i / N), c = (int)(i % N);
    d[(long long)c * K + k] = (__bf16)s[i];
  }
}

// ---------------------------------------------------------------- QKV GEMM
// A: xb [kM][kC] bf16 row-major.  Bt: wqkvT [3C][kC] bf16 (N-major).
// Per wave: 32x64 output tile, K-step 32, 8 wmmas / step, direct global frags.
__global__ void __launch_bounds__(256)
gemm_qkv(const __bf16* __restrict__ A, const __bf16* __restrict__ Bt,
         const float* __restrict__ bias,
         __bf16* __restrict__ Qo, __bf16* __restrict__ Ko,
         __bf16* __restrict__ Vto) {
  const int lane = threadIdx.x & 31;
  const int lo = lane & 15, hi = lane >> 4;
  const int wid = blockIdx.x * (blockDim.x >> 5) + (threadIdx.x >> 5);
  const int NT = kNqkv / 64;                 // 48
  const int mt = wid / NT, nt = wid % NT;
  if (mt >= kM / 32) return;
  const int m0 = mt * 32, n0 = nt * 64;

  const __bf16* ap0 = A + (size_t)(m0 + lo) * kC + hi * 8;
  const __bf16* ap1 = A + (size_t)(m0 + 16 + lo) * kC + hi * 8;
  const __bf16* bp0 = Bt + (size_t)(n0 + lo) * kC + hi * 8;

  v8f acc[2][4];
#pragma unroll
  for (int mi = 0; mi < 2; ++mi)
#pragma unroll
    for (int ni = 0; ni < 4; ++ni) acc[mi][ni] = v8f{0.f,0.f,0.f,0.f,0.f,0.f,0.f,0.f};

#pragma unroll 2
  for (int kk = 0; kk < kC; kk += 32) {
    Frag16 a[2], b[4];
    a[0].u[0] = ld16(ap0 + kk);
    a[0].u[1] = ld16(ap0 + kk + 16);
    a[1].u[0] = ld16(ap1 + kk);
    a[1].u[1] = ld16(ap1 + kk + 16);
#pragma unroll
    for (int ni = 0; ni < 4; ++ni) {
      const __bf16* bp = bp0 + (size_t)ni * 16 * kC + kk;
      b[ni].u[0] = ld16(bp);
      b[ni].u[1] = ld16(bp + 16);
    }
#pragma unroll
    for (int mi = 0; mi < 2; ++mi)
#pragma unroll
      for (int ni = 0; ni < 4; ++ni)
        acc[mi][ni] = wmma_bf16(a[mi].bf, b[ni].bf, acc[mi][ni]);
  }

  // epilogue: +bias, scatter Q,K as [B,H,T,D]; V transposed as [B,H,D,T]
#pragma unroll
  for (int mi = 0; mi < 2; ++mi)
#pragma unroll
    for (int ni = 0; ni < 4; ++ni)
#pragma unroll
      for (int r = 0; r < 8; ++r) {
        const int m = m0 + mi * 16 + hi * 8 + r;
        const int n = n0 + ni * 16 + lo;
        const float v = acc[mi][ni][r] + bias[n];
        const int bb = m / kT, t = m % kT;
        const int which = n / kC, c = n % kC;
        const int h = c / kD, d = c % kD;
        const size_t bh = (size_t)(bb * kH + h);
        const __bf16 bv = (__bf16)v;
        if (which == 0)      Qo[(bh * kT + t) * kD + d] = bv;
        else if (which == 1) Ko[(bh * kT + t) * kD + d] = bv;
        else                 Vto[(bh * kD + d) * kT + t] = bv;
      }
}

// ---------------------------------------------------------------- attention
// One wave = 32 queries of one (b,h) (two 16-query N-tiles sharing all K/V
// fragments -> half the L2 traffic per FLOP). Computes S^T = K.Q^T so the
// softmaxed probabilities repack in-lane into the A-fragment for P.V.
__global__ void __launch_bounds__(256)
attn(const __bf16* __restrict__ Q, const __bf16* __restrict__ K,
     const __bf16* __restrict__ Vt, __bf16* __restrict__ Y) {
  const int lane = threadIdx.x & 31;
  const int lo = lane & 15, hi = lane >> 4;
  const int wid = blockIdx.x * (blockDim.x >> 5) + (threadIdx.x >> 5);
  const int QTILES = kT / 32;                 // 64
  const int qtile = wid % QTILES;
  const int bh = wid / QTILES;                // 0..63
  if (bh >= kB * kH) return;
  const int h = bh % kH, bb = bh / kH;
  const int q0 = qtile * 32;

  const __bf16* Qp = Q + (size_t)bh * kT * kD;
  const __bf16* Kp = K + (size_t)bh * kT * kD;
  const __bf16* Vp = Vt + (size_t)bh * kD * kT;

  // Q^T B-fragments [qn][ks]: N = query = lo, two 8-d chunks at ks*32+8*hi and +16
  Frag16 qf[2][2];
#pragma unroll
  for (int qn = 0; qn < 2; ++qn)
#pragma unroll
    for (int ks = 0; ks < 2; ++ks) {
      const __bf16* qp = Qp + (size_t)(q0 + qn * 16 + lo) * kD + ks * 32 + hi * 8;
      qf[qn][ks].u[0] = ld16(qp);
      qf[qn][ks].u[1] = ld16(qp + 16);
    }

  v8f O[2][4];
#pragma unroll
  for (int qn = 0; qn < 2; ++qn)
#pragma unroll
    for (int ni = 0; ni < 4; ++ni) O[qn][ni] = v8f{0.f,0.f,0.f,0.f,0.f,0.f,0.f,0.f};
  float m_run[2] = {-1e30f, -1e30f};
  float l_run[2] = {0.f, 0.f};

  const int kend = q0 + 32;                   // causal: keys < q0+32
  for (int k0 = 0; k0 < kend; k0 += 32) {
    // K A-fragments [mi][ks]: 16 key rows x 32 d, shared by both query tiles
    Frag16 a[2][2];
#pragma unroll
    for (int mi = 0; mi < 2; ++mi)
#pragma unroll
      for (int ks = 0; ks < 2; ++ks) {
        const __bf16* kp = Kp + (size_t)(k0 + mi * 16 + lo) * kD + ks * 32 + hi * 8;
        a[mi][ks].u[0] = ld16(kp);
        a[mi][ks].u[1] = ld16(kp + 16);
      }
    if (k0 + 32 < kend)                       // L2/L0 prefetch of next K block
      __builtin_prefetch(Kp + (size_t)(k0 + 32 + lane) * kD, 0, 1);

    // S^T tiles [qn][mi]: rows = keys, cols = queries
    v8f st[2][2];
#pragma unroll
    for (int qn = 0; qn < 2; ++qn)
#pragma unroll
      for (int mi = 0; mi < 2; ++mi) {
        st[qn][mi] = v8f{0.f,0.f,0.f,0.f,0.f,0.f,0.f,0.f};
#pragma unroll
        for (int ks = 0; ks < 2; ++ks)
          st[qn][mi] = wmma_bf16(a[mi][ks].bf, qf[qn][ks].bf, st[qn][mi]);
      }

    // mask + online softmax in place (per-query column, query = q0+qn*16+lo)
    float scale[2];
#pragma unroll
    for (int qn = 0; qn < 2; ++qn) {
      const int qq = q0 + qn * 16 + lo;
      float mb = -1e30f;
#pragma unroll
      for (int mi = 0; mi < 2; ++mi)
#pragma unroll
        for (int r = 0; r < 8; ++r) {
          const int key = k0 + mi * 16 + hi * 8 + r;
          float s = st[qn][mi][r] * 0.125f;   // 1/sqrt(64)
          s = (key <= qq) ? s : -1e30f;
          st[qn][mi][r] = s;
          mb = fmaxf(mb, s);
        }
      mb = fmaxf(mb, __shfl_xor(mb, 16, 32));
      const float m_new = fmaxf(m_run[qn], mb);
      scale[qn] = __expf(m_run[qn] - m_new);
      float ls = 0.f;
#pragma unroll
      for (int mi = 0; mi < 2; ++mi)
#pragma unroll
        for (int r = 0; r < 8; ++r) {
          const float e = __expf(st[qn][mi][r] - m_new);
          st[qn][mi][r] = e;
          ls += e;
        }
      ls += __shfl_xor(ls, 16, 32);
      l_run[qn] = l_run[qn] * scale[qn] + ls;
      m_run[qn] = m_new;
    }

    // rescale O rows (row = query hi*8+r; stats live at lane (query&15))
#pragma unroll
    for (int qn = 0; qn < 2; ++qn) {
      float srow[8];
#pragma unroll
      for (int r = 0; r < 8; ++r) srow[r] = __shfl(scale[qn], hi * 8 + r, 32);
#pragma unroll
      for (int ni = 0; ni < 4; ++ni)
#pragma unroll
        for (int r = 0; r < 8; ++r) O[qn][ni][r] *= srow[r];
    }

    // in-lane repack: S^T C-frags -> P A-frags (16 q x 32 k each)
    Frag16 pf[2];
#pragma unroll
    for (int qn = 0; qn < 2; ++qn)
#pragma unroll
      for (int e = 0; e < 8; ++e) {
        pf[qn].bf[e]     = (__bf16)st[qn][0][e];
        pf[qn].bf[8 + e] = (__bf16)st[qn][1][e];
      }

    // O += P.V  (V^T rows give keys 8*hi+0..7, 16+8*hi+0..7; V frags shared by both qn)
#pragma unroll
    for (int ni = 0; ni < 4; ++ni) {
      Frag16 vf;
      const __bf16* vp = Vp + (size_t)(ni * 16 + lo) * kT + k0 + hi * 8;
      vf.u[0] = ld16(vp);
      vf.u[1] = ld16(vp + 16);
#pragma unroll
      for (int qn = 0; qn < 2; ++qn)
        O[qn][ni] = wmma_bf16(pf[qn].bf, vf.bf, O[qn][ni]);
    }
  }

  // normalize and store to y [B,T,C] bf16
#pragma unroll
  for (int qn = 0; qn < 2; ++qn) {
    float linv[8];
#pragma unroll
    for (int r = 0; r < 8; ++r) linv[r] = 1.0f / __shfl(l_run[qn], hi * 8 + r, 32);
#pragma unroll
    for (int ni = 0; ni < 4; ++ni)
#pragma unroll
      for (int r = 0; r < 8; ++r) {
        const int qq2 = q0 + qn * 16 + hi * 8 + r;
        const int dcol = ni * 16 + lo;
        Y[((size_t)bb * kT + qq2) * kC + h * kD + dcol] =
            (__bf16)(O[qn][ni][r] * linv[r]);
      }
  }
}

// ---------------------------------------------------------------- proj GEMM
__global__ void __launch_bounds__(256)
gemm_proj(const __bf16* __restrict__ A, const __bf16* __restrict__ Bt,
          const float* __restrict__ bias, float* __restrict__ out) {
  const int lane = threadIdx.x & 31;
  const int lo = lane & 15, hi = lane >> 4;
  const int wid = blockIdx.x * (blockDim.x >> 5) + (threadIdx.x >> 5);
  const int NT = kC / 64;                    // 16
  const int mt = wid / NT, nt = wid % NT;
  if (mt >= kM / 32) return;
  const int m0 = mt * 32, n0 = nt * 64;

  const __bf16* ap0 = A + (size_t)(m0 + lo) * kC + hi * 8;
  const __bf16* ap1 = A + (size_t)(m0 + 16 + lo) * kC + hi * 8;
  const __bf16* bp0 = Bt + (size_t)(n0 + lo) * kC + hi * 8;

  v8f acc[2][4];
#pragma unroll
  for (int mi = 0; mi < 2; ++mi)
#pragma unroll
    for (int ni = 0; ni < 4; ++ni) acc[mi][ni] = v8f{0.f,0.f,0.f,0.f,0.f,0.f,0.f,0.f};

#pragma unroll 2
  for (int kk = 0; kk < kC; kk += 32) {
    Frag16 a[2], b[4];
    a[0].u[0] = ld16(ap0 + kk);
    a[0].u[1] = ld16(ap0 + kk + 16);
    a[1].u[0] = ld16(ap1 + kk);
    a[1].u[1] = ld16(ap1 + kk + 16);
#pragma unroll
    for (int ni = 0; ni < 4; ++ni) {
      const __bf16* bp = bp0 + (size_t)ni * 16 * kC + kk;
      b[ni].u[0] = ld16(bp);
      b[ni].u[1] = ld16(bp + 16);
    }
#pragma unroll
    for (int mi = 0; mi < 2; ++mi)
#pragma unroll
      for (int ni = 0; ni < 4; ++ni)
        acc[mi][ni] = wmma_bf16(a[mi].bf, b[ni].bf, acc[mi][ni]);
  }

#pragma unroll
  for (int mi = 0; mi < 2; ++mi)
#pragma unroll
    for (int ni = 0; ni < 4; ++ni)
#pragma unroll
      for (int r = 0; r < 8; ++r) {
        const int m = m0 + mi * 16 + hi * 8 + r;
        const int n = n0 + ni * 16 + lo;
        out[(size_t)m * kC + n] = acc[mi][ni][r] + bias[n];
      }
}

// ---------------------------------------------------------------- launcher
extern "C" void kernel_launch(void* const* d_in, const int* in_sizes, int n_in,
                              void* d_out, int out_size, void* d_ws, size_t ws_size,
                              hipStream_t stream) {
  const float* x      = (const float*)d_in[0];   // [B,T,C]
  const float* w_qkv  = (const float*)d_in[1];   // [C,3C]
  const float* b_qkv  = (const float*)d_in[2];   // [3C]
  const float* w_proj = (const float*)d_in[3];   // [C,C]
  const float* b_proj = (const float*)d_in[4];   // [C]
  float* out = (float*)d_out;

  char* p = (char*)d_ws;
  __bf16* xb     = (__bf16*)p; p += (size_t)kM * kC * 2;       // 16 MB
  __bf16* wqkvT  = (__bf16*)p; p += (size_t)kNqkv * kC * 2;    //  6 MB
  __bf16* wprojT = (__bf16*)p; p += (size_t)kC * kC * 2;       //  2 MB
  __bf16* qb     = (__bf16*)p; p += (size_t)kM * kC * 2;       // 16 MB
  __bf16* kb     = (__bf16*)p; p += (size_t)kM * kC * 2;       // 16 MB
  __bf16* vt     = (__bf16*)p; p += (size_t)kM * kC * 2;       // 16 MB
  __bf16* yb     = xb;  // x is dead after the QKV GEMM; reuse for attn output

  (void)in_sizes; (void)n_in; (void)out_size; (void)ws_size;

  cvt_f32_bf16<<<2048, 256, 0, stream>>>(x, xb, (long long)kM * kC);
  transpose_f32_bf16<<<2048, 256, 0, stream>>>(w_qkv, wqkvT, kC, kNqkv);
  transpose_f32_bf16<<<1024, 256, 0, stream>>>(w_proj, wprojT, kC, kC);

  // QKV GEMM: (8192/32)*(3072/64) = 12288 waves, 8 waves/block
  gemm_qkv<<<1536, 256, 0, stream>>>(xb, wqkvT, b_qkv, qb, kb, vt);

  // attention: 4*16*(2048/32) = 4096 waves
  attn<<<512, 256, 0, stream>>>(qb, kb, vt, yb);

  // proj GEMM: (8192/32)*(1024/64) = 4096 waves
  gemm_proj<<<512, 256, 0, stream>>>(yb, wprojT, b_proj, out);
}
